// VisionTransformer_58342835749176
// MI455X (gfx1250) — matrix operand order repaired and checked by reading, not compiled, not run on hardware
//
#include <hip/hip_runtime.h>
#include <cstdint>
#include <cstddef>

// ---------------------------------------------------------------------------
// MI455X (gfx1250) ViT forward.  Compute-bound (~2 TFLOP vs ~0.5 GB traffic)
// -> every matmul via v_wmma_f32_16x16x32_bf16; global->LDS via CDNA5 async
//    copies (ASYNCcnt) with double-buffered tiles.
// ---------------------------------------------------------------------------

typedef __bf16 bf16;
typedef bf16  v16bf __attribute__((ext_vector_type(16)));
typedef float v8f   __attribute__((ext_vector_type(8)));

__device__ __forceinline__ bf16  f2b(float x) { return (bf16)x; }

__device__ __forceinline__ v8f vzero8() {
  v8f v;
  #pragma unroll
  for (int i = 0; i < 8; ++i) v[i] = 0.0f;
  return v;
}

__device__ __forceinline__ v8f wmma_bf16(v16bf a, v16bf b, v8f c) {
  return __builtin_amdgcn_wmma_f32_16x16x32_bf16(
      false, a, false, b, (short)0, c, false, false);
}

// A-fragment K index (16-bit A 16x32, ISA 7.12.2): two contiguous 8-elem runs
__device__ __forceinline__ int a_kidx(int e, int hi) {
  return (e & 7) + (hi << 3) + ((e >> 3) << 4);
}

// CDNA5 async global->LDS copy (16B per lane), tracked by ASYNCcnt.
__device__ __forceinline__ uint32_t lds_off(const void* p) {
  return (uint32_t)(uintptr_t)p;  // low 32 bits of flat shared addr = LDS offset
}
__device__ __forceinline__ void async_load_b128(uint32_t dst_lds,
                                                const void* gsrc) {
  asm volatile("global_load_async_to_lds_b128 %0, %1, off"
               :: "v"(dst_lds), "v"((uint64_t)(uintptr_t)gsrc)
               : "memory");
}
__device__ __forceinline__ void wait_async() {
  asm volatile("s_wait_asynccnt 0" ::: "memory");
}

// ---------------------------------------------------------------------------
// Generic bf16 WMMA GEMM:  C[M,N] = A[M,K] * Wt  where Wt is [K,N] bf16
// (weights pre-transposed).  128x128 block tile, 256 threads = 8 waves
// (2M x 4N), 4x2 accum tiles per wave, double-buffered async LDS tiles.
// ---------------------------------------------------------------------------
#define ASTR  40   // LDS stride for A tiles   [128][40]
#define WSTRN 136  // LDS stride for Wt tiles  [32][136]

__global__ __launch_bounds__(256) void gemm_bf16_wmma(
    const bf16* __restrict__ A,      // [M,K] row-major
    const bf16* __restrict__ Wt,     // [K,N] row-major (pre-transposed weights)
    const float* __restrict__ bias,  // [N] or nullptr
    const float* __restrict__ resid, // [M,N] or nullptr
    float* __restrict__ Cf,          // [M,N] f32 out or nullptr
    bf16* __restrict__ Cb,           // [M,N] bf16 out or nullptr
    int M, int N, int K, int gelu) {
  __shared__ bf16 As[2][128 * ASTR];
  __shared__ bf16 Ws[2][32 * WSTRN];

  const int tid  = threadIdx.x;
  const int lane = tid & 31;
  const int wave = tid >> 5;
  const int lr   = lane & 15;
  const int hi   = lane >> 4;
  const int wm   = wave >> 2;  // 0..1
  const int wn   = wave & 3;   // 0..3

  const int rowBase = blockIdx.y * 128;
  const int colBase = blockIdx.x * 128;

  v8f acc[4][2];
  #pragma unroll
  for (int mt = 0; mt < 4; ++mt)
    #pragma unroll
    for (int nt = 0; nt < 2; ++nt) acc[mt][nt] = vzero8();

  auto load_tiles = [&](int k0, int bufi) {
    // A tile 128x32: 512 chunks of 8 bf16
    #pragma unroll
    for (int i = 0; i < 2; ++i) {
      int idx = tid + i * 256;
      int r   = idx >> 2;
      int cs  = (idx & 3) * 8;
      async_load_b128(lds_off(&As[bufi][r * ASTR + cs]),
                      A + (size_t)(rowBase + r) * K + k0 + cs);
    }
    // Wt tile 32x128: 512 chunks of 8 bf16
    #pragma unroll
    for (int i = 0; i < 2; ++i) {
      int idx = tid + i * 256;
      int kk  = idx >> 4;
      int cs  = (idx & 15) * 8;
      async_load_b128(lds_off(&Ws[bufi][kk * WSTRN + cs]),
                      Wt + (size_t)(k0 + kk) * N + colBase + cs);
    }
  };

  load_tiles(0, 0);
  wait_async();
  __syncthreads();

  int buf = 0;
  for (int k0 = 0; k0 < K; k0 += 32) {
    if (k0 + 32 < K) load_tiles(k0 + 32, buf ^ 1);

    v16bf af[4], bfrag[2];
    #pragma unroll
    for (int mt = 0; mt < 4; ++mt) {
      int row = wm * 64 + mt * 16 + lr;
      #pragma unroll
      for (int e = 0; e < 16; ++e)
        af[mt][e] = As[buf][row * ASTR + a_kidx(e, hi)];
    }
    #pragma unroll
    for (int nt = 0; nt < 2; ++nt) {
      int cb = wn * 32 + nt * 16;
      #pragma unroll
      for (int e = 0; e < 16; ++e)
        bfrag[nt][e] = Ws[buf][a_kidx(e, hi) * WSTRN + cb + lr];  // k = a_kidx, n = lr
    }
    #pragma unroll
    for (int mt = 0; mt < 4; ++mt)
      #pragma unroll
      for (int nt = 0; nt < 2; ++nt)
        acc[mt][nt] = wmma_bf16(af[mt], bfrag[nt], acc[mt][nt]);

    wait_async();
    __syncthreads();
    buf ^= 1;
  }

  #pragma unroll
  for (int mt = 0; mt < 4; ++mt)
    #pragma unroll
    for (int nt = 0; nt < 2; ++nt)
      #pragma unroll
      for (int r = 0; r < 8; ++r) {
        int row = rowBase + wm * 64 + mt * 16 + hi * 8 + r;
        int col = colBase + wn * 32 + nt * 16 + lr;
        float v = acc[mt][nt][r];
        if (bias) v += bias[col];
        if (gelu) v = 0.5f * v * (1.0f + erff(v * 0.70710678118654752f));
        if (resid) v += resid[(size_t)row * N + col];
        if (Cf) Cf[(size_t)row * N + col] = v;
        if (Cb) Cb[(size_t)row * N + col] = f2b(v);
      }
}

// ---------------------------------------------------------------------------
// Flash attention (per batch, head, 64-row q-block; 16 x 64-key blocks).
// K stored transposed in LDS (Kt[d][key]) so S-fragments are contiguous.
// Double-buffered K/V with async V loads; next K held in registers.
// ---------------------------------------------------------------------------
#define QROWS 64
#define KKEYS 64
#define QSTR  136  // bf16 stride for 128-wide rows (Q, V)
#define KTSTR 72   // bf16 stride for Kt rows (64 keys + pad)
#define SSTR  68   // f32 stride for S
#define PSTR  72   // bf16 stride for P
#define NKB   16   // 1024 / 64 key blocks

__global__ __launch_bounds__(256) void attn_flash_wmma(
    const bf16* __restrict__ qkv,  // [B*n, 3*D] bf16
    bf16* __restrict__ outb,       // [B*n, D] bf16
    float scale) {
  const int n = 1024, Dm = 1024, hd = 128;
  const int qb = blockIdx.x, hidx = blockIdx.y, b = blockIdx.z;

  extern __shared__ char smem[];
  bf16*  Qs  = (bf16*)smem;                    // [64][136]
  bf16*  Kt0 = Qs + QROWS * QSTR;              // [2][128][72]
  bf16*  Vs0 = Kt0 + 2 * 128 * KTSTR;          // [2][64][136]
  float* Ss  = (float*)(Vs0 + 2 * KKEYS * QSTR);  // [64][68]
  bf16*  Ps  = (bf16*)(Ss + QROWS * SSTR);     // [64][72]
  float* rm  = (float*)(Ps + QROWS * PSTR);    // [64]
  float* rl  = rm + QROWS;                     // [64]
  float* rc  = rl + QROWS;                     // [64]
  float* pred = rc + QROWS;                    // [256] partials

  const int tid  = threadIdx.x;
  const int lane = tid & 31;
  const int wave = tid >> 5;
  const int lr   = lane & 15;
  const int hi   = lane >> 4;
  const int mrow = 16 * (wave >> 1);
  const int half = wave & 1;

  // async-load Q block (64 x 128)
  const bf16* qbase = qkv + (size_t)(b * n + qb * QROWS) * 3072 + hidx * hd;
  #pragma unroll
  for (int i = 0; i < 4; ++i) {
    int idx = tid + i * 256;
    int r   = idx >> 4;
    int cs  = (idx & 15) * 8;
    async_load_b128(lds_off(&Qs[r * QSTR + cs]),
                    qbase + (size_t)r * 3072 + cs);
  }
  if (tid < QROWS) { rm[tid] = -3.0e30f; rl[tid] = 0.0f; }

  v8f oacc[4];
  #pragma unroll
  for (int t = 0; t < 4; ++t) oacc[t] = vzero8();

  // prologue: K/V block 0  (V async, K transposed via regular stores)
  {
    const bf16* kbp = qkv + (size_t)(b * n) * 3072 + Dm + hidx * hd;
    #pragma unroll
    for (int i = 0; i < 4; ++i) {
      int idx = tid + i * 256;
      int r   = idx >> 4;
      int cs  = (idx & 15) * 8;
      async_load_b128(lds_off(&Vs0[r * QSTR + cs]),
                      kbp + Dm + (size_t)r * 3072 + cs);
    }
    #pragma unroll
    for (int i = 0; i < 8; ++i) {
      int idx = tid + i * 256;
      int r   = idx >> 5;          // key 0..63
      int cs  = (idx & 31) * 4;    // d 0..124
      union { uint2 u; bf16 h[4]; } cv;
      cv.u = *(const uint2*)(kbp + (size_t)r * 3072 + cs);
      #pragma unroll
      for (int j = 0; j < 4; ++j) Kt0[(cs + j) * KTSTR + r] = cv.h[j];
    }
  }
  wait_async();
  __syncthreads();

  int cur = 0;
  for (int kb = 0; kb < NKB; ++kb) {
    bf16* Kt  = Kt0 + cur * 128 * KTSTR;
    bf16* Vs  = Vs0 + cur * KKEYS * QSTR;
    bf16* Ktn = Kt0 + (cur ^ 1) * 128 * KTSTR;
    bf16* Vsn = Vs0 + (cur ^ 1) * KKEYS * QSTR;

    const bool hasNext = (kb + 1 < NKB);
    uint2 kreg[8];
    const bf16* kbn =
        qkv + (size_t)(b * n + (kb + 1) * KKEYS) * 3072 + Dm + hidx * hd;
    if (hasNext) {
      #pragma unroll
      for (int i = 0; i < 4; ++i) {
        int idx = tid + i * 256;
        int r   = idx >> 4;
        int cs  = (idx & 15) * 8;
        async_load_b128(lds_off(&Vsn[r * QSTR + cs]),
                        kbn + Dm + (size_t)r * 3072 + cs);
      }
      #pragma unroll
      for (int i = 0; i < 8; ++i) {
        int idx = tid + i * 256;
        int r   = idx >> 5;
        int cs  = (idx & 31) * 4;
        kreg[i] = *(const uint2*)(kbn + (size_t)r * 3072 + cs);
      }
    }

    // S = Q K^T * scale : wave -> rows mrow..+15, key cols half*32..+32
    {
      const int ncol = 32 * half;
      v8f s0 = vzero8(), s1 = vzero8();
      #pragma unroll
      for (int kk = 0; kk < 4; ++kk) {  // hd/32
        v16bf a;
        #pragma unroll
        for (int e = 0; e < 16; ++e)
          a[e] = Qs[(mrow + lr) * QSTR + kk * 32 + a_kidx(e, hi)];
        v16bf b0, b1;
        #pragma unroll
        for (int e = 0; e < 16; ++e) {
          b0[e] = Kt[(kk * 32 + a_kidx(e, hi)) * KTSTR + ncol + lr];       // k = a_kidx, n = lr
          b1[e] = Kt[(kk * 32 + a_kidx(e, hi)) * KTSTR + ncol + 16 + lr];  // k = a_kidx, n = lr
        }
        s0 = wmma_bf16(a, b0, s0);
        s1 = wmma_bf16(a, b1, s1);
      }
      #pragma unroll
      for (int r = 0; r < 8; ++r) {
        int row = mrow + r + 8 * hi;
        Ss[row * SSTR + ncol + lr]      = s0[r] * scale;
        Ss[row * SSTR + ncol + 16 + lr] = s1[r] * scale;
      }
    }
    __syncthreads();

    // online softmax, 4 threads per row
    const int row = tid >> 2, q = tid & 3;
    {
      float lm = -3.0e30f;
      #pragma unroll
      for (int c = 0; c < 16; ++c)
        lm = fmaxf(lm, Ss[row * SSTR + q * 16 + c]);
      pred[tid] = lm;
    }
    __syncthreads();
    if (q == 0) {
      float m_old = rm[row];
      float mx = fmaxf(fmaxf(pred[row * 4 + 0], pred[row * 4 + 1]),
                       fmaxf(pred[row * 4 + 2], pred[row * 4 + 3]));
      mx = fmaxf(mx, m_old);
      rm[row] = mx;
      rc[row] = __expf(m_old - mx);
    }
    __syncthreads();
    {
      float mx = rm[row];
      float ps = 0.0f;
      #pragma unroll
      for (int c = 0; c < 16; ++c) {
        float pv = __expf(Ss[row * SSTR + q * 16 + c] - mx);
        ps += pv;
        Ps[row * PSTR + q * 16 + c] = f2b(pv);
      }
      pred[tid] = ps;
    }
    __syncthreads();
    if (q == 0)
      rl[row] = rl[row] * rc[row] + pred[row * 4 + 0] + pred[row * 4 + 1] +
                pred[row * 4 + 2] + pred[row * 4 + 3];

    // O = rc*O + P @ V
    {
      const int dcol = 64 * half;
      #pragma unroll
      for (int t = 0; t < 4; ++t)
        #pragma unroll
        for (int r8 = 0; r8 < 8; ++r8)
          oacc[t][r8] *= rc[mrow + r8 + 8 * hi];
      #pragma unroll
      for (int kk = 0; kk < 2; ++kk) {  // 64 keys / 32
        v16bf a;
        #pragma unroll
        for (int e = 0; e < 16; ++e)
          a[e] = Ps[(mrow + lr) * PSTR + kk * 32 + a_kidx(e, hi)];
        #pragma unroll
        for (int t = 0; t < 4; ++t) {
          v16bf bv;
          #pragma unroll
          for (int e = 0; e < 16; ++e)
            bv[e] = Vs[(kk * 32 + a_kidx(e, hi)) * QSTR + dcol + t * 16 + lr];
          oacc[t] = wmma_bf16(a, bv, oacc[t]);
        }
      }
    }

    // store next K tile (transposed) from registers
    if (hasNext) {
      #pragma unroll
      for (int i = 0; i < 8; ++i) {
        int idx = tid + i * 256;
        int r   = idx >> 5;
        int cs  = (idx & 31) * 4;
        union { uint2 u; bf16 h[4]; } cv;
        cv.u = kreg[i];
        #pragma unroll
        for (int j = 0; j < 4; ++j) Ktn[(cs + j) * KTSTR + r] = cv.h[j];
      }
    }
    wait_async();
    __syncthreads();
    cur ^= 1;
  }

  // normalize + store
  {
    const int dcol = 64 * half;
    #pragma unroll
    for (int t = 0; t < 4; ++t)
      #pragma unroll
      for (int r8 = 0; r8 < 8; ++r8) {
        int row = mrow + r8 + 8 * hi;
        float v = oacc[t][r8] / rl[row];
        outb[(size_t)(b * n + qb * QROWS + row) * Dm + hidx * hd + dcol +
             t * 16 + lr] = f2b(v);
      }
  }
}

// ---------------------------------------------------------------------------
// Support kernels
// ---------------------------------------------------------------------------

// batched tiled transpose+convert: src [L][N][K] f32 -> dst [L][K][N] bf16
__global__ __launch_bounds__(256) void transpose_cvt(
    const float* __restrict__ src, bf16* __restrict__ dst, int N, int K) {
  __shared__ bf16 tile[32][33];
  const float* s = src + (size_t)blockIdx.z * N * K;
  bf16* d = dst + (size_t)blockIdx.z * N * K;
  const int kb = blockIdx.x * 32, nb = blockIdx.y * 32;
  const int tx = threadIdx.x & 31, tg = threadIdx.x >> 5;
  #pragma unroll
  for (int i = 0; i < 4; ++i) {
    int y = tg * 4 + i;  // n within tile
    tile[y][tx] = f2b(s[(size_t)(nb + y) * K + kb + tx]);
  }
  __syncthreads();
  #pragma unroll
  for (int i = 0; i < 4; ++i) {
    int y = tg * 4 + i;  // k within tile
    d[(size_t)(kb + y) * N + nb + tx] = tile[tx][y];
  }
}

// im2col: x[B,1,512,512] f32 -> patches [B*1024, 256] bf16
__global__ void im2col_patches(const float* __restrict__ x,
                               bf16* __restrict__ patches) {
  size_t idx = (size_t)blockIdx.x * blockDim.x + threadIdx.x;
  if (idx >= (size_t)8 * 1024 * 256) return;
  int c = (int)(idx & 255);
  int t = (int)((idx >> 8) & 1023);
  int b = (int)(idx >> 18);
  int row = ((t >> 5) << 4) + (c >> 4);
  int col = ((t & 31) << 4) + (c & 15);
  patches[idx] = f2b(x[((size_t)b << 18) + ((size_t)row << 9) + col]);
}

__global__ void add_pos_enc(float* __restrict__ tok) {
  size_t idx = (size_t)blockIdx.x * blockDim.x + threadIdx.x;
  if (idx >= (size_t)8 * 1024 * 1024) return;
  int d = (int)(idx & 1023);
  int t = (int)((idx >> 10) & 1023);
  float freq = __powf(10000.0f, -(float)(2 * (d & ~1)) * (1.0f / 1024.0f));
  float ang  = (float)t * freq;
  tok[idx] += (d & 1) ? __cosf(ang) : __sinf(ang);
}

__global__ __launch_bounds__(256) void layernorm_bf16(
    const float* __restrict__ x, const float* __restrict__ g,
    const float* __restrict__ bb, bf16* __restrict__ y) {
  __shared__ float r1[256];
  __shared__ float r2[256];
  const int tid = threadIdx.x;
  const size_t row = blockIdx.x;
  const float4 v4 = *(const float4*)(x + row * 1024 + tid * 4);
  float s  = v4.x + v4.y + v4.z + v4.w;
  float s2 = v4.x * v4.x + v4.y * v4.y + v4.z * v4.z + v4.w * v4.w;
  r1[tid] = s; r2[tid] = s2;
  __syncthreads();
  for (int o = 128; o > 0; o >>= 1) {
    if (tid < o) { r1[tid] += r1[tid + o]; r2[tid] += r2[tid + o]; }
    __syncthreads();
  }
  float mean = r1[0] * (1.0f / 1024.0f);
  float var  = r2[0] * (1.0f / 1024.0f) - mean * mean;
  float inv  = rsqrtf(var + 1e-5f);
  const float vv[4] = {v4.x, v4.y, v4.z, v4.w};
  #pragma unroll
  for (int i = 0; i < 4; ++i) {
    int d = tid * 4 + i;
    y[row * 1024 + d] = f2b((vv[i] - mean) * inv * g[d] + bb[d]);
  }
}

__global__ void reconstruct(const float* __restrict__ tok,
                            float* __restrict__ out) {
  size_t idx = (size_t)blockIdx.x * blockDim.x + threadIdx.x;
  if (idx >= (size_t)8 * 1024 * 1024) return;
  int x = (int)(idx & 1023);
  int y = (int)((idx >> 10) & 1023);
  int b = (int)(idx >> 20);
  int t = ((y >> 5) << 5) + (x >> 5);
  int d = ((((y >> 1) & 15) << 4) + ((x >> 1) & 15)) * 4 + ((y & 1) << 1) +
          (x & 1);
  out[idx] = tok[(((size_t)b << 10) + t) * 1024 + d];
}

// ---------------------------------------------------------------------------
// Host orchestration
// ---------------------------------------------------------------------------
extern "C" void kernel_launch(void* const* d_in, const int* in_sizes, int n_in,
                              void* d_out, int out_size, void* d_ws,
                              size_t ws_size, hipStream_t stream) {
  (void)in_sizes; (void)n_in; (void)out_size; (void)ws_size;
  const float* x      = (const float*)d_in[0];
  const float* conv_w = (const float*)d_in[1];
  const float* conv_b = (const float*)d_in[2];
  const float* ln1_g  = (const float*)d_in[3];
  const float* ln1_b  = (const float*)d_in[4];
  const float* qkv_w  = (const float*)d_in[5];
  const float* qkv_b  = (const float*)d_in[6];
  const float* proj_w = (const float*)d_in[7];
  const float* proj_b = (const float*)d_in[8];
  const float* ln2_g  = (const float*)d_in[9];
  const float* ln2_b  = (const float*)d_in[10];
  const float* fc1_w  = (const float*)d_in[11];
  const float* fc1_b  = (const float*)d_in[12];
  const float* fc2_w  = (const float*)d_in[13];
  const float* fc2_b  = (const float*)d_in[14];
  float* out = (float*)d_out;

  const int M = 8192, Dm = 1024;

  char* p = (char*)d_ws;
  auto carve = [&](size_t bytes) -> char* {
    char* r = p;
    p += (bytes + 255) & ~(size_t)255;
    return r;
  };
  bf16* wconvT = (bf16*)carve((size_t)1024 * 256 * 2);           // [256][1024]
  bf16* wqkvT  = (bf16*)carve((size_t)12 * 3072 * 1024 * 2);     // [1024][3072]
  bf16* wprojT = (bf16*)carve((size_t)12 * 1024 * 1024 * 2);     // [1024][1024]
  bf16* wfc1T  = (bf16*)carve((size_t)12 * 2048 * 1024 * 2);     // [1024][2048]
  bf16* wfc2T  = (bf16*)carve((size_t)12 * 1024 * 2048 * 2);     // [2048][1024]
  bf16* patches = (bf16*)carve((size_t)M * 256 * 2);
  float* tok  = (float*)carve((size_t)M * 1024 * 4);
  bf16* hbuf  = (bf16*)carve((size_t)M * 1024 * 2);
  bf16* qkvb  = (bf16*)carve((size_t)M * 3072 * 2);
  bf16* ubuf  = (bf16*)carve((size_t)M * 2048 * 2);

  // pre-transpose weights to [K,N] bf16 (per layer)
  transpose_cvt<<<dim3(256 / 32, 1024 / 32, 1),  256, 0, stream>>>(conv_w, wconvT, 1024, 256);
  transpose_cvt<<<dim3(1024 / 32, 3072 / 32, 12), 256, 0, stream>>>(qkv_w, wqkvT, 3072, 1024);
  transpose_cvt<<<dim3(1024 / 32, 1024 / 32, 12), 256, 0, stream>>>(proj_w, wprojT, 1024, 1024);
  transpose_cvt<<<dim3(1024 / 32, 2048 / 32, 12), 256, 0, stream>>>(fc1_w, wfc1T, 2048, 1024);
  transpose_cvt<<<dim3(2048 / 32, 1024 / 32, 12), 256, 0, stream>>>(fc2_w, wfc2T, 1024, 2048);

  // patch embed + positional encoding
  im2col_patches<<<(unsigned)(((size_t)M * 256 + 255) / 256), 256, 0, stream>>>(
      x, patches);
  gemm_bf16_wmma<<<dim3(Dm / 128, M / 128), 256, 0, stream>>>(
      patches, wconvT, conv_b, nullptr, tok, nullptr, M, Dm, 256, 0);
  add_pos_enc<<<(unsigned)(((size_t)M * 1024 + 255) / 256), 256, 0, stream>>>(
      tok);

  const float scale = 0.08838834764831845f;  // 1/sqrt(128)
  const size_t attn_smem =
      (size_t)QROWS * QSTR * 2 +          // Qs
      (size_t)2 * 128 * KTSTR * 2 +       // Kt x2
      (size_t)2 * KKEYS * QSTR * 2 +      // Vs x2
      (size_t)QROWS * SSTR * 4 +          // Ss
      (size_t)QROWS * PSTR * 2 +          // Ps
      (size_t)3 * QROWS * 4 +             // rm/rl/rc
      (size_t)256 * 4;                    // pred

  for (int l = 0; l < 12; ++l) {
    layernorm_bf16<<<M, 256, 0, stream>>>(tok, ln1_g + (size_t)l * 1024,
                                          ln1_b + (size_t)l * 1024, hbuf);
    gemm_bf16_wmma<<<dim3(3072 / 128, M / 128), 256, 0, stream>>>(
        hbuf, wqkvT + (size_t)l * 3072 * 1024, qkv_b + (size_t)l * 3072,
        nullptr, nullptr, qkvb, M, 3072, 1024, 0);
    attn_flash_wmma<<<dim3(1024 / QROWS, 8, 8), 256, attn_smem, stream>>>(
        qkvb, hbuf, scale);
    gemm_bf16_wmma<<<dim3(Dm / 128, M / 128), 256, 0, stream>>>(
        hbuf, wprojT + (size_t)l * 1024 * 1024, proj_b + (size_t)l * 1024, tok,
        tok, nullptr, M, Dm, 1024, 0);
    layernorm_bf16<<<M, 256, 0, stream>>>(tok, ln2_g + (size_t)l * 1024,
                                          ln2_b + (size_t)l * 1024, hbuf);
    gemm_bf16_wmma<<<dim3(2048 / 128, M / 128), 256, 0, stream>>>(
        hbuf, wfc1T + (size_t)l * 2048 * 1024, fc1_b + (size_t)l * 2048,
        nullptr, nullptr, ubuf, M, 2048, 1024, 1);
    gemm_bf16_wmma<<<dim3(Dm / 128, M / 128), 256, 0, stream>>>(
        ubuf, wfc2T + (size_t)l * 1024 * 2048, fc2_b + (size_t)l * 1024, tok,
        tok, nullptr, M, Dm, 2048, 0);
  }

  reconstruct<<<(unsigned)(((size_t)8 * 1024 * 1024 + 255) / 256), 256, 0,
                stream>>>(tok, out);
}
